// CNNLSTMANN_33689723470073
// MI455X (gfx1250) — compile-verified, matrix operand order checked by reading, not run on hardware
//
#include <hip/hip_runtime.h>
#include <hip/hip_bf16.h>
#include <math.h>

// ---------------------------------------------------------------------------
// Model dims
// ---------------------------------------------------------------------------
#define BB   128          // batch
#define TT   60           // timesteps
#define L0   100          // raw length
#define HH   256          // hidden
#define G4   1024         // 4*H
#define NBT  (BB*TT)      // 7680

typedef __attribute__((ext_vector_type(16))) _Float16 v16h;
typedef __attribute__((ext_vector_type(8)))  float    v8f;

// ---------------------------------------------------------------------------
// Small helpers
// ---------------------------------------------------------------------------
__device__ __forceinline__ float sigm(float x) { return 1.0f / (1.0f + expf(-x)); }

// Load a 16x32 f16 tile in the CDNA5 WMMA A/B register layout.
// src is row-major [rows x ld] f16; tile rows row0..row0+15, cols k0..k0+31.
// Lanes 0-15 : row = row0+lane,     K = {k0+0..7, k0+16..23}
// Lanes 16-31: row = row0+lane-16,  K = {k0+8..15, k0+24..31}
__device__ __forceinline__ v16h ld_tile16(const _Float16* __restrict__ src,
                                          int ld, int row0, int k0, int lane) {
    int r     = row0 + (lane & 15);
    int kbase = k0 + ((lane & 16) ? 8 : 0);
    const _Float16* p = src + (long)r * ld + kbase;
    v16h a;
#pragma unroll
    for (int j = 0; j < 8; ++j) a[j] = p[j];
#pragma unroll
    for (int j = 0; j < 8; ++j) a[8 + j] = p[16 + j];
    return a;
}

// ---------------------------------------------------------------------------
// Conv / BN / pool stage (scalar fp32 — tiny fraction of total work)
// ---------------------------------------------------------------------------
__global__ void conv1d_relu(const float* __restrict__ in, const float* __restrict__ w,
                            const float* __restrict__ bias, float* __restrict__ out,
                            int N, int Cin, int Lin, int Cout, int Lout, int stride) {
    long idx = (long)blockIdx.x * 256 + threadIdx.x;
    long total = (long)N * Cout * Lout;
    if (idx >= total) return;
    int l  = (int)(idx % Lout);
    int co = (int)((idx / Lout) % Cout);
    int n  = (int)(idx / ((long)Cout * Lout));
    float acc = bias[co];
    for (int ci = 0; ci < Cin; ++ci) {
        const float* ip = in + ((long)n * Cin + ci) * Lin;
        const float* wp = w + ((long)co * Cin + ci) * 3;
#pragma unroll
        for (int j = 0; j < 3; ++j) {
            int li = l * stride - 1 + j;
            if (li >= 0 && li < Lin) acc += ip[li] * wp[j];
        }
    }
    out[idx] = fmaxf(acc, 0.0f);
}

// BN per (t,c): stats over batch (128) and length L. x is [B*T, C, L], row=b*T+t.
__global__ void bn_per_step(float* __restrict__ x, const float* __restrict__ g,
                            const float* __restrict__ b, int C, int L) {
    __shared__ float rs[256], rq[256];
    __shared__ float s_scale, s_shift;
    int t = blockIdx.x / C;
    int c = blockIdx.x % C;
    int tid = threadIdx.x;
    int count = BB * L;
    float sum = 0.0f, sq = 0.0f;
    for (int i = tid; i < count; i += 256) {
        int bb = i / L, l = i % L;
        float v = x[(((long)bb * TT + t) * C + c) * L + l];
        sum += v; sq += v * v;
    }
    rs[tid] = sum; rq[tid] = sq;
    __syncthreads();
    for (int s = 128; s > 0; s >>= 1) {
        if (tid < s) { rs[tid] += rs[tid + s]; rq[tid] += rq[tid + s]; }
        __syncthreads();
    }
    if (tid == 0) {
        float mean = rs[0] / count;
        float var  = rq[0] / count - mean * mean;
        float sc   = rsqrtf(var + 1e-5f) * g[c];
        s_scale = sc;
        s_shift = b[c] - mean * sc;
    }
    __syncthreads();
    float sc = s_scale, sh = s_shift;
    for (int i = tid; i < count; i += 256) {
        int bb = i / L, l = i % L;
        long a = (((long)bb * TT + t) * C + c) * L + l;
        x[a] = x[a] * sc + sh;
    }
}

__global__ void maxpool_k(const float* __restrict__ in, float* __restrict__ out,
                          int N, int C, int Lin, int k, int Lout) {
    long idx = (long)blockIdx.x * 256 + threadIdx.x;
    long total = (long)N * C * Lout;
    if (idx >= total) return;
    int l = (int)(idx % Lout);
    int c = (int)((idx / Lout) % C);
    int n = (int)(idx / ((long)C * Lout));
    const float* ip = in + ((long)n * C + c) * Lin + l * k;
    float m = ip[0];
    for (int j = 1; j < k; ++j) m = fmaxf(m, ip[j]);
    out[idx] = m;
}

// ---------------------------------------------------------------------------
// Utility kernels
// ---------------------------------------------------------------------------
__global__ void f32_to_f16_k(const float* __restrict__ in, _Float16* __restrict__ out, int n) {
    int i = blockIdx.x * 256 + threadIdx.x;
    if (i < n) out[i] = (_Float16)in[i];
}
__global__ void zero_f32_k(float* __restrict__ p, int n) {
    int i = blockIdx.x * 256 + threadIdx.x;
    if (i < n) p[i] = 0.0f;
}
__global__ void bias_sum_k(const float* __restrict__ a, const float* __restrict__ b,
                           float* __restrict__ out) {
    int i = blockIdx.x * 256 + threadIdx.x;
    if (i < G4) out[i] = a[i] + b[i];
}

// ---------------------------------------------------------------------------
// LSTM input projection: xp[m, g] = bsum[g] + sum_k x16[m,k] * wih16[g,k]
// m in [0,7680), g in [0,1024). One 16x16 tile per wave, K-loop of 8 WMMAs.
// grid = 3840 blocks x 256 threads (8 waves/block), 30720 tiles.
// ---------------------------------------------------------------------------
__global__ void lstm_xp_gemm(const _Float16* __restrict__ x16,
                             const _Float16* __restrict__ wih16,
                             const float* __restrict__ bsum,
                             float* __restrict__ xp) {
    int wave = blockIdx.x * 8 + (threadIdx.x >> 5);
    int lane = threadIdx.x & 31;
    int m0 = (wave >> 6) * 16;   // 480 row tiles
    int n0 = (wave & 63) * 16;   // 64 col tiles
    int n  = n0 + (lane & 15);
    float bv = bsum[n];
    v8f acc;
#pragma unroll
    for (int r = 0; r < 8; ++r) acc[r] = bv;
#pragma unroll
    for (int kk = 0; kk < 8; ++kk) {
        int k0 = kk * 32;
        v16h a  = ld_tile16(x16,   HH, m0, k0, lane);
        v16h bm = ld_tile16(wih16, HH, n0, k0, lane);
        acc = __builtin_amdgcn_wmma_f32_16x16x32_f16(false, a, false, bm,
                                                     (short)0, acc, false, false);
    }
#pragma unroll
    for (int r = 0; r < 8; ++r) {
        int m = m0 + r + ((lane & 16) ? 8 : 0);
        xp[(long)m * G4 + n] = acc[r];
    }
}

// ---------------------------------------------------------------------------
// Fused recurrent step: gates = xp_t + h·whhᵀ ; then cell/hidden update.
// One wave owns a 16(batch) x 16(hidden) patch and all 4 gates for it.
// grid = 16 blocks x 256 threads = 128 waves (8 m-tiles x 16 n-tiles).
// h double-buffered in f16 (read old, write new); c updated in place.
// ---------------------------------------------------------------------------
__global__ void lstm_step(const _Float16* __restrict__ h16_in,
                          _Float16* __restrict__ h16_out,
                          const _Float16* __restrict__ whh16,
                          const float* __restrict__ xp_t,   // + t*1024, row stride T*1024
                          float* __restrict__ cbuf,         // [128,256]
                          float* __restrict__ hseq_t) {     // + t*256, row stride T*256
    int wave = blockIdx.x * 8 + (threadIdx.x >> 5);
    int lane = threadIdx.x & 31;
    int m0 = (wave >> 4) * 16;   // batch tile (0..7)
    int n0 = (wave & 15) * 16;   // hidden tile (0..15)
    int n  = n0 + (lane & 15);

    v8f ai, af, ag, ao;
#pragma unroll
    for (int r = 0; r < 8; ++r) {
        int b = m0 + r + ((lane & 16) ? 8 : 0);
        const float* xr = xp_t + (long)b * (TT * G4);
        ai[r] = xr[0 * HH + n];
        af[r] = xr[1 * HH + n];
        ag[r] = xr[2 * HH + n];
        ao[r] = xr[3 * HH + n];
    }
#pragma unroll
    for (int kk = 0; kk < 8; ++kk) {
        int k0 = kk * 32;
        v16h a = ld_tile16(h16_in, HH, m0, k0, lane);
        v16h b0 = ld_tile16(whh16, HH, 0 * HH + n0, k0, lane);
        ai = __builtin_amdgcn_wmma_f32_16x16x32_f16(false, a, false, b0, (short)0, ai, false, false);
        v16h b1 = ld_tile16(whh16, HH, 1 * HH + n0, k0, lane);
        af = __builtin_amdgcn_wmma_f32_16x16x32_f16(false, a, false, b1, (short)0, af, false, false);
        v16h b2 = ld_tile16(whh16, HH, 2 * HH + n0, k0, lane);
        ag = __builtin_amdgcn_wmma_f32_16x16x32_f16(false, a, false, b2, (short)0, ag, false, false);
        v16h b3 = ld_tile16(whh16, HH, 3 * HH + n0, k0, lane);
        ao = __builtin_amdgcn_wmma_f32_16x16x32_f16(false, a, false, b3, (short)0, ao, false, false);
    }
#pragma unroll
    for (int r = 0; r < 8; ++r) {
        int b = m0 + r + ((lane & 16) ? 8 : 0);
        float iv = sigm(ai[r]);
        float fv = sigm(af[r]);
        float gv = tanhf(ag[r]);
        float ov = sigm(ao[r]);
        long ci = (long)b * HH + n;
        float cn = fv * cbuf[ci] + iv * gv;
        cbuf[ci] = cn;
        float hn = ov * tanhf(cn);
        hseq_t[(long)b * (TT * HH) + n] = hn;
        h16_out[ci] = (_Float16)hn;
    }
}

// ---------------------------------------------------------------------------
// MLP head: h[b] (256) -> 128 -> 32 -> 1 -> sigmoid. One block per batch row.
// ---------------------------------------------------------------------------
__global__ void fc_head(const float* __restrict__ hseq,  // [B, T, 256]
                        const float* __restrict__ w1, const float* __restrict__ b1,
                        const float* __restrict__ w2, const float* __restrict__ b2,
                        const float* __restrict__ w3, const float* __restrict__ b3,
                        float* __restrict__ out) {
    __shared__ float h[256];
    __shared__ float s1[128];
    __shared__ float s2[32];
    int b = blockIdx.x, tid = threadIdx.x;
    h[tid] = hseq[(long)b * (TT * HH) + 59 * HH + tid];
    __syncthreads();
    if (tid < 128) {
        float s = b1[tid];
        for (int k = 0; k < 256; ++k) s += w1[tid * 256 + k] * h[k];
        s1[tid] = s;
    }
    __syncthreads();
    if (tid < 32) {
        float s = b2[tid];
        for (int k = 0; k < 128; ++k) s += w2[tid * 128 + k] * s1[k];
        s2[tid] = s;
    }
    __syncthreads();
    if (tid == 0) {
        float s = b3[0];
        for (int k = 0; k < 32; ++k) s += w3[k] * s2[k];
        out[b] = 1.0f / (1.0f + expf(-s));
    }
}

// ---------------------------------------------------------------------------
// Host-side orchestration
// ---------------------------------------------------------------------------
extern "C" void kernel_launch(void* const* d_in, const int* in_sizes, int n_in,
                              void* d_out, int out_size, void* d_ws, size_t ws_size,
                              hipStream_t stream) {
    (void)in_sizes; (void)n_in; (void)out_size; (void)ws_size;
    const float* wave    = (const float*)d_in[0];
    const float* c1w = (const float*)d_in[1];  const float* c1b = (const float*)d_in[2];
    const float* c2w = (const float*)d_in[3];  const float* c2b = (const float*)d_in[4];
    const float* c3w = (const float*)d_in[5];  const float* c3b = (const float*)d_in[6];
    const float* c4w = (const float*)d_in[7];  const float* c4b = (const float*)d_in[8];
    const float* bn1g = (const float*)d_in[9];  const float* bn1b = (const float*)d_in[10];
    const float* bn2g = (const float*)d_in[11]; const float* bn2b = (const float*)d_in[12];
    const float* bn3g = (const float*)d_in[13]; const float* bn3b = (const float*)d_in[14];
    const float* bn4g = (const float*)d_in[15]; const float* bn4b = (const float*)d_in[16];
    const float* wih = (const float*)d_in[17];  // (5,1024,256)
    const float* whh = (const float*)d_in[18];  // (5,1024,256)
    const float* bih = (const float*)d_in[19];  // (5,1024)
    const float* bhh = (const float*)d_in[20];
    const float* l1w = (const float*)d_in[21]; const float* l1b = (const float*)d_in[22];
    const float* l2w = (const float*)d_in[23]; const float* l2b = (const float*)d_in[24];
    const float* l3w = (const float*)d_in[25]; const float* l3b = (const float*)d_in[26];
    float* out = (float*)d_out;

    // Workspace regions (floats). Peak ~327 MB.
    float* ws = (float*)d_ws;
    float* rA = ws;                   // 49,152,000 f
    float* rB = ws + 49152000;        // 24,576,000 f
    float* rC = ws + 73728000;        //  7,864,320 f

    // ---- Conv front-end ----
    // conv1: (7680,1,100) -> (7680,64,100) in rA
    conv1d_relu<<<192000, 256, 0, stream>>>(wave, c1w, c1b, rA, NBT, 1, 100, 64, 100, 1);
    bn_per_step<<<TT * 64, 256, 0, stream>>>(rA, bn1g, bn1b, 64, 100);
    // conv2 stride2: -> (7680,64,50) in rB
    conv1d_relu<<<96000, 256, 0, stream>>>(rA, c2w, c2b, rB, NBT, 64, 100, 64, 50, 2);
    bn_per_step<<<TT * 64, 256, 0, stream>>>(rB, bn2g, bn2b, 64, 50);
    // maxpool k=3: -> (7680,64,16) in rC
    maxpool_k<<<30720, 256, 0, stream>>>(rB, rC, NBT, 64, 50, 3, 16);
    // conv3: -> (7680,128,16) in rA
    conv1d_relu<<<61440, 256, 0, stream>>>(rC, c3w, c3b, rA, NBT, 64, 16, 128, 16, 1);
    bn_per_step<<<TT * 128, 256, 0, stream>>>(rA, bn3g, bn3b, 128, 16);
    // conv4 stride2: -> (7680,256,8) in rB
    conv1d_relu<<<61440, 256, 0, stream>>>(rA, c4w, c4b, rB, NBT, 128, 16, 256, 8, 2);
    bn_per_step<<<TT * 256, 256, 0, stream>>>(rB, bn4g, bn4b, 256, 8);
    // maxpool k=5: -> (7680,256) in rC  == LSTM input (B,T,256)
    maxpool_k<<<7680, 256, 0, stream>>>(rB, rC, NBT, 256, 8, 5, 1);

    // ---- LSTM scratch (inside rA, free after conv4) ----
    float*     xp     = rA;                                   // 7,864,320 f
    float*     hseqA  = rA + 7864320;                         // 1,966,080 f
    float*     hseqB  = rA + 9830400;                         // 1,966,080 f
    _Float16*  x16    = (_Float16*)(rA + 11796480);           // 1,966,080 halves
    _Float16*  wih16  = (_Float16*)(rA + 12779520);           //   262,144 halves
    _Float16*  whh16  = (_Float16*)(rA + 12910592);
    float*     bsum   = rA + 13041664;                        // 1024 f
    _Float16*  h16a   = (_Float16*)(rA + 13042688);           // 32,768 halves
    _Float16*  h16b   = (_Float16*)(rA + 13059072);
    float*     cbuf   = rA + 13075456;                        // 32,768 f

    for (int l = 0; l < 5; ++l) {
        const float* xin = (l == 0) ? rC : ((l & 1) ? hseqA : hseqB);
        float* hseq_out  = (l & 1) ? hseqB : hseqA;

        f32_to_f16_k<<<1024, 256, 0, stream>>>(wih + (long)l * G4 * HH, wih16, G4 * HH);
        f32_to_f16_k<<<1024, 256, 0, stream>>>(whh + (long)l * G4 * HH, whh16, G4 * HH);
        bias_sum_k<<<4, 256, 0, stream>>>(bih + l * G4, bhh + l * G4, bsum);
        f32_to_f16_k<<<7680, 256, 0, stream>>>(xin, x16, NBT * HH);

        // xp = x·wihᵀ + (bih+bhh)  [7680 x 1024]
        lstm_xp_gemm<<<3840, 256, 0, stream>>>(x16, wih16, bsum, xp);

        zero_f32_k<<<128, 256, 0, stream>>>(cbuf, BB * HH);
        zero_f32_k<<<64, 256, 0, stream>>>((float*)h16a, BB * HH / 2);

        for (int t = 0; t < TT; ++t) {
            const _Float16* hin = (t & 1) ? h16b : h16a;
            _Float16*       hou = (t & 1) ? h16a : h16b;
            lstm_step<<<16, 256, 0, stream>>>(hin, hou, whh16,
                                              xp + (long)t * G4, cbuf,
                                              hseq_out + (long)t * HH);
        }
    }

    // Layer 4 output landed in hseqA.
    fc_head<<<BB, 256, 0, stream>>>(hseqA, l1w, l1b, l2w, l2b, l3w, l3b, out);
}
